// NonLocalBlock_28252294873533
// MI455X (gfx1250) — compile-verified, matrix operand order checked by reading, not run on hardware
//
#include <hip/hip_runtime.h>
#include <hip/hip_bf16.h>
#include <math.h>

// ---------------------------------------------------------------------------
// Problem constants (from the reference)
// ---------------------------------------------------------------------------
#define Bb 16
#define Hh 64
#define Ww 64
#define Cc 192
#define HEAD_DIM 32
#define N_HEADS 6
#define Pp 8
#define SHIFT 4
#define NW 64            // (H/P)*(W/P) windows per image
#define PP 64            // P*P tokens per window
#define RTOK (Bb*Hh*Ww)  // 65536 token rows
#define QK_SCALE 0.17677669529663687f  // 1/sqrt(32)

typedef __attribute__((ext_vector_type(16))) __bf16 v16bf;
typedef __attribute__((ext_vector_type(8)))  float  v8f;

__device__ __forceinline__ v8f vzero8() {
  v8f z;
  #pragma unroll
  for (int i = 0; i < 8; ++i) z[i] = 0.0f;
  return z;
}

// Load one 16-bit WMMA fragment (A or B operand) for this lane.
// ISA layout (16-bit 16x32 A / 32x16 B): lane half selects kbase {0,8};
// elements 0..7 = K[kbase..kbase+7], elements 8..15 = K[kbase+16..kbase+23].
// `p` must point at (row_base + kbase) and be 16B aligned.
__device__ __forceinline__ v16bf load_frag16(const __bf16* p) {
  v16bf f;
  uint4 lo = *reinterpret_cast<const uint4*>(p);
  uint4 hi = *reinterpret_cast<const uint4*>(p + 16);
  reinterpret_cast<uint4*>(&f)[0] = lo;
  reinterpret_cast<uint4*>(&f)[1] = hi;
  return f;
}

// ---------------------------------------------------------------------------
// Shared GEMM core: C(16x64 per wave) += A[row,:K] * BT[col,:K]^T in bf16 WMMA.
// Block = 256 thr = 8 waves; block tile = 128 rows x 64 cols.
// A: [M,K] bf16 row-major.  BT: [N,K] bf16 row-major (i.e. W transposed).
// ---------------------------------------------------------------------------
template <int KK>
__device__ __forceinline__ void gemm_core(const __bf16* __restrict__ A,
                                          const __bf16* __restrict__ BT,
                                          int row, int col0, int lane,
                                          v8f acc[4]) {
  const int kbase = (lane >> 4) * 8;
  const int nlo   = lane & 15;
  const __bf16* arow = A + (size_t)row * KK + kbase;
  #pragma unroll
  for (int k0 = 0; k0 < KK; k0 += 32) {
    v16bf a = load_frag16(arow + k0);
    if (k0 + 32 < KK) __builtin_prefetch(arow + k0 + 32, 0, 3);  // global_prefetch_b8
    #pragma unroll
    for (int nt = 0; nt < 4; ++nt) {
      v16bf b = load_frag16(BT + (size_t)(col0 + nt * 16 + nlo) * KK + k0 + kbase);
      acc[nt] = __builtin_amdgcn_wmma_f32_16x16x32_bf16(
          false, a, false, b, (short)0, acc[nt], false, false);
    }
  }
}

// ---------------------------------------------------------------------------
// Small prep kernels
// ---------------------------------------------------------------------------

// W [K,N] f32 -> WT [N,K] bf16
__global__ void k_transpose_bf16(const float* __restrict__ src,
                                 __bf16* __restrict__ dst, int K, int N) {
  int idx = blockIdx.x * 256 + threadIdx.x;
  if (idx >= K * N) return;
  int n = idx / K, k = idx - n * K;
  dst[idx] = (__bf16)src[(size_t)k * N + n];
}

// rel_pos [6,15,15] -> relb [6,64,64]
__global__ void k_relbias(const float* __restrict__ rel_pos,
                          float* __restrict__ relb) {
  int idx = blockIdx.x * 256 + threadIdx.x;
  if (idx >= N_HEADS * PP * PP) return;
  int hd = idx >> 12, pq = idx & 4095, p = pq >> 6, q = pq & 63;
  int dy = (p >> 3) - (q >> 3) + Pp - 1;
  int dx = (p & 7) - (q & 7) + Pp - 1;
  relb[idx] = rel_pos[hd * (2 * Pp - 1) * (2 * Pp - 1) + dy * (2 * Pp - 1) + dx];
}

// ---------------------------------------------------------------------------
// LN1 + roll(-SHIFT) + window partition.  One wave per token (32 lanes x 6 ch).
// Output rows are in window order: r = ((b*NW)+win)*PP + p
// ---------------------------------------------------------------------------
__global__ __launch_bounds__(256) void k_ln1_window(
    const float* __restrict__ x, const float* __restrict__ g,
    const float* __restrict__ bia, __bf16* __restrict__ h) {
  const int lane = threadIdx.x & 31;
  const int token = blockIdx.x * 8 + (threadIdx.x >> 5);
  const int bb = token >> 12;
  const int win = (token >> 6) & 63;
  const int p = token & 63;
  const int wy = win >> 3, wx = win & 7, py = p >> 3, px = p & 7;
  const int sh = (wy * Pp + py + SHIFT) & (Hh - 1);
  const int sw = (wx * Pp + px + SHIFT) & (Ww - 1);
  const float* src = x + (((size_t)bb * Hh + sh) * Ww + sw) * Cc;
  float v[6], s = 0.f, ss = 0.f;
  #pragma unroll
  for (int i = 0; i < 6; ++i) {
    v[i] = src[lane + 32 * i];
    s += v[i]; ss += v[i] * v[i];
  }
  #pragma unroll
  for (int m = 16; m >= 1; m >>= 1) {
    s  += __shfl_xor(s,  m, 32);
    ss += __shfl_xor(ss, m, 32);
  }
  const float mu = s * (1.f / Cc);
  const float rs = rsqrtf(ss * (1.f / Cc) - mu * mu + 1e-5f);
  __bf16* dst = h + (size_t)token * Cc;
  #pragma unroll
  for (int i = 0; i < 6; ++i) {
    int c = lane + 32 * i;
    dst[c] = (__bf16)((v[i] - mu) * rs * g[c] + bia[c]);
  }
}

// LN2 over natural (b,h,w) token order; input x1 f32 in ws.
__global__ __launch_bounds__(256) void k_ln2(
    const float* __restrict__ x1, const float* __restrict__ g,
    const float* __restrict__ bia, __bf16* __restrict__ h2) {
  const int lane = threadIdx.x & 31;
  const int token = blockIdx.x * 8 + (threadIdx.x >> 5);
  const float* src = x1 + (size_t)token * Cc;
  float v[6], s = 0.f, ss = 0.f;
  #pragma unroll
  for (int i = 0; i < 6; ++i) {
    v[i] = src[lane + 32 * i];
    s += v[i]; ss += v[i] * v[i];
  }
  #pragma unroll
  for (int m = 16; m >= 1; m >>= 1) {
    s  += __shfl_xor(s,  m, 32);
    ss += __shfl_xor(ss, m, 32);
  }
  const float mu = s * (1.f / Cc);
  const float rs = rsqrtf(ss * (1.f / Cc) - mu * mu + 1e-5f);
  __bf16* dst = h2 + (size_t)token * Cc;
  #pragma unroll
  for (int i = 0; i < 6; ++i) {
    int c = lane + 32 * i;
    dst[c] = (__bf16)((v[i] - mu) * rs * g[c] + bia[c]);
  }
}

// ---------------------------------------------------------------------------
// QKV GEMM: [65536,192] x [192,576] + bias.  Q scaled by 1/sqrt(d) on store;
// Q,K stored [bn,head,p,d]; V stored transposed [bn,head,d,p] for P*V WMMA.
// ---------------------------------------------------------------------------
__global__ __launch_bounds__(256) void k_gemm_qkv(
    const __bf16* __restrict__ Hm, const __bf16* __restrict__ WT,
    const float* __restrict__ bqkv, __bf16* __restrict__ Q,
    __bf16* __restrict__ Kd, __bf16* __restrict__ Vt) {
  const int lane = threadIdx.x & 31, wave = threadIdx.x >> 5;
  const int row = blockIdx.x * 128 + wave * 16 + (lane & 15);
  const int col0 = blockIdx.y * 64;
  v8f acc[4];
  #pragma unroll
  for (int t = 0; t < 4; ++t) acc[t] = vzero8();
  gemm_core<Cc>(Hm, WT, row, col0, lane, acc);

  const int half = lane >> 4, nlo = lane & 15;
  #pragma unroll
  for (int nt = 0; nt < 4; ++nt) {
    const int n = col0 + nt * 16 + nlo;
    const float bias = bqkv[n];
    const int which = n / Cc;
    const int nn = n - which * Cc;
    const int head = nn >> 5, d = nn & 31;
    #pragma unroll
    for (int j = 0; j < 8; ++j) {
      const int r = blockIdx.x * 128 + wave * 16 + j + 8 * half;
      const int bn = r >> 6, p = r & 63;
      const float v = acc[nt][j] + bias;
      if (which == 0)
        Q[(((size_t)bn * N_HEADS + head) * PP + p) * HEAD_DIM + d] =
            (__bf16)(v * QK_SCALE);
      else if (which == 1)
        Kd[(((size_t)bn * N_HEADS + head) * PP + p) * HEAD_DIM + d] = (__bf16)v;
      else
        Vt[(((size_t)bn * N_HEADS + head) * HEAD_DIM + d) * PP + p] = (__bf16)v;
    }
  }
}

// ---------------------------------------------------------------------------
// Attention: one block (4 waves) per (window, head). Wave = 16 rows of S.
// S = Q*K^T (one WMMA K-step since HEAD_DIM==32), +rel_bias, +shift mask,
// softmax (shfl within 16-lane halves), then P*V via WMMA (P staged in LDS
// to convert C-layout -> A-fragment layout).
// ---------------------------------------------------------------------------
__global__ __launch_bounds__(128) void k_attn(
    const __bf16* __restrict__ Q, const __bf16* __restrict__ Kd,
    const __bf16* __restrict__ Vt, const float* __restrict__ relb,
    __bf16* __restrict__ attn) {
  __shared__ __align__(16) __bf16 Plds[4][16][PP];  // 8 KB
  const int lane = threadIdx.x & 31, wave = threadIdx.x >> 5;
  const int bn = blockIdx.x, head = blockIdx.y;
  const int kbase = (lane >> 4) * 8, nlo = lane & 15, half = lane >> 4;
  const __bf16* Qb  = Q  + ((size_t)bn * N_HEADS + head) * PP * HEAD_DIM;
  const __bf16* Kb  = Kd + ((size_t)bn * N_HEADS + head) * PP * HEAD_DIM;
  const __bf16* Vtb = Vt + ((size_t)bn * N_HEADS + head) * HEAD_DIM * PP;

  // --- S = Q K^T : 4 WMMA per wave (16x64 strip), single K-step (d=32)
  v16bf qf = load_frag16(Qb + (size_t)(wave * 16 + nlo) * HEAD_DIM + kbase);
  v8f s[4];
  #pragma unroll
  for (int t = 0; t < 4; ++t) {
    v16bf kf = load_frag16(Kb + (size_t)(t * 16 + nlo) * HEAD_DIM + kbase);
    s[t] = __builtin_amdgcn_wmma_f32_16x16x32_bf16(
        false, qf, false, kf, (short)0, vzero8(), false, false);
  }

  // --- bias + shift mask (computed inline; only boundary windows masked)
  const int win = bn & (NW - 1);
  const int wy = win >> 3, wx = win & 7;
  float rowv[4][8];
  #pragma unroll
  for (int t = 0; t < 4; ++t) {
    const int q = t * 16 + nlo;
    const int qy = q >> 3, qx = q & 7;
    #pragma unroll
    for (int j = 0; j < 8; ++j) {
      const int p = wave * 16 + j + 8 * half;
      const int py = p >> 3, px = p & 7;
      float v = s[t][j] + relb[((size_t)head * PP + p) * PP + q];
      const bool msk = (wy == 7 && ((py < SHIFT) != (qy < SHIFT))) ||
                       (wx == 7 && ((px < SHIFT) != (qx < SHIFT)));
      rowv[t][j] = msk ? -INFINITY : v;
    }
  }

  // --- softmax over q (64 cols live across the lane's 16-lane half x 4 tiles)
  float sum[8];
  #pragma unroll
  for (int j = 0; j < 8; ++j) {
    float m = rowv[0][j];
    #pragma unroll
    for (int t = 1; t < 4; ++t) m = fmaxf(m, rowv[t][j]);
    #pragma unroll
    for (int d = 8; d >= 1; d >>= 1) m = fmaxf(m, __shfl_xor(m, d, 32));
    float ssum = 0.f;
    #pragma unroll
    for (int t = 0; t < 4; ++t) {
      rowv[t][j] = __expf(rowv[t][j] - m);
      ssum += rowv[t][j];
    }
    #pragma unroll
    for (int d = 8; d >= 1; d >>= 1) ssum += __shfl_xor(ssum, d, 32);
    sum[j] = 1.0f / ssum;
  }

  // --- stage normalized P into LDS (per-wave tile; same-wave DS is in-order)
  #pragma unroll
  for (int t = 0; t < 4; ++t) {
    #pragma unroll
    for (int j = 0; j < 8; ++j)
      Plds[wave][j + 8 * half][t * 16 + nlo] = (__bf16)(rowv[t][j] * sum[j]);
  }

  // --- O = P V : A-fragments from LDS, B-fragments from V^T (contiguous)
  v8f o[2];
  o[0] = vzero8(); o[1] = vzero8();
  #pragma unroll
  for (int ks = 0; ks < 2; ++ks) {
    v16bf pf = load_frag16(&Plds[wave][nlo][ks * 32 + kbase]);
    #pragma unroll
    for (int t2 = 0; t2 < 2; ++t2) {
      v16bf vf = load_frag16(Vtb + (size_t)(t2 * 16 + nlo) * PP + ks * 32 + kbase);
      o[t2] = __builtin_amdgcn_wmma_f32_16x16x32_bf16(
          false, pf, false, vf, (short)0, o[t2], false, false);
    }
  }

  // --- store attn in [r,192] row-major (heads concatenated) for proj GEMM
  #pragma unroll
  for (int t2 = 0; t2 < 2; ++t2) {
    const int d = t2 * 16 + nlo;
    #pragma unroll
    for (int j = 0; j < 8; ++j) {
      const int p = wave * 16 + j + 8 * half;
      attn[((size_t)bn * PP + p) * Cc + head * HEAD_DIM + d] = (__bf16)o[t2][j];
    }
  }
}

// ---------------------------------------------------------------------------
// Proj GEMM + inverse window/roll + residual -> x1 (f32)
// ---------------------------------------------------------------------------
__global__ __launch_bounds__(256) void k_gemm_proj(
    const __bf16* __restrict__ A, const __bf16* __restrict__ WT,
    const float* __restrict__ bproj, const float* __restrict__ x,
    float* __restrict__ x1) {
  const int lane = threadIdx.x & 31, wave = threadIdx.x >> 5;
  const int row = blockIdx.x * 128 + wave * 16 + (lane & 15);
  const int col0 = blockIdx.y * 64;
  v8f acc[4];
  #pragma unroll
  for (int t = 0; t < 4; ++t) acc[t] = vzero8();
  gemm_core<Cc>(A, WT, row, col0, lane, acc);

  const int half = lane >> 4, nlo = lane & 15;
  #pragma unroll
  for (int nt = 0; nt < 4; ++nt) {
    const int n = col0 + nt * 16 + nlo;
    const float bias = bproj[n];
    #pragma unroll
    for (int j = 0; j < 8; ++j) {
      const int r = blockIdx.x * 128 + wave * 16 + j + 8 * half;
      const int bn = r >> 6, p = r & 63;
      const int b = bn >> 6, win = bn & 63;
      const int wy = win >> 3, wx = win & 7, py = p >> 3, px = p & 7;
      const int oh = (wy * Pp + py + SHIFT) & (Hh - 1);   // inverse roll(+SHIFT)
      const int ow = (wx * Pp + px + SHIFT) & (Ww - 1);
      const size_t idx = (((size_t)b * Hh + oh) * Ww + ow) * Cc + n;
      x1[idx] = x[idx] + acc[nt][j] + bias;
    }
  }
}

// ---------------------------------------------------------------------------
// MLP1 GEMM + exact GELU -> bf16
// ---------------------------------------------------------------------------
__global__ __launch_bounds__(256) void k_gemm_mlp1(
    const __bf16* __restrict__ A, const __bf16* __restrict__ WT,
    const float* __restrict__ bm1, __bf16* __restrict__ M1) {
  const int lane = threadIdx.x & 31, wave = threadIdx.x >> 5;
  const int row = blockIdx.x * 128 + wave * 16 + (lane & 15);
  const int col0 = blockIdx.y * 64;
  v8f acc[4];
  #pragma unroll
  for (int t = 0; t < 4; ++t) acc[t] = vzero8();
  gemm_core<Cc>(A, WT, row, col0, lane, acc);

  const int half = lane >> 4, nlo = lane & 15;
  #pragma unroll
  for (int nt = 0; nt < 4; ++nt) {
    const int n = col0 + nt * 16 + nlo;
    const float bias = bm1[n];
    #pragma unroll
    for (int j = 0; j < 8; ++j) {
      const int r = blockIdx.x * 128 + wave * 16 + j + 8 * half;
      float v = acc[nt][j] + bias;
      v = 0.5f * v * (1.0f + erff(v * 0.70710678118654752f));  // exact GELU
      M1[(size_t)r * (4 * Cc) + n] = (__bf16)v;
    }
  }
}

// ---------------------------------------------------------------------------
// MLP2 GEMM (K=768) + residual -> f32 output
// ---------------------------------------------------------------------------
__global__ __launch_bounds__(256) void k_gemm_mlp2(
    const __bf16* __restrict__ A, const __bf16* __restrict__ WT,
    const float* __restrict__ bm2, const float* __restrict__ x1,
    float* __restrict__ out) {
  const int lane = threadIdx.x & 31, wave = threadIdx.x >> 5;
  const int row = blockIdx.x * 128 + wave * 16 + (lane & 15);
  const int col0 = blockIdx.y * 64;
  v8f acc[4];
  #pragma unroll
  for (int t = 0; t < 4; ++t) acc[t] = vzero8();
  gemm_core<4 * Cc>(A, WT, row, col0, lane, acc);

  const int half = lane >> 4, nlo = lane & 15;
  #pragma unroll
  for (int nt = 0; nt < 4; ++nt) {
    const int n = col0 + nt * 16 + nlo;
    const float bias = bm2[n];
    #pragma unroll
    for (int j = 0; j < 8; ++j) {
      const int r = blockIdx.x * 128 + wave * 16 + j + 8 * half;
      const size_t idx = (size_t)r * Cc + n;
      out[idx] = x1[idx] + acc[nt][j] + bias;
    }
  }
}

// ---------------------------------------------------------------------------
// Host-side orchestration
// ---------------------------------------------------------------------------
extern "C" void kernel_launch(void* const* d_in, const int* in_sizes, int n_in,
                              void* d_out, int out_size, void* d_ws, size_t ws_size,
                              hipStream_t stream) {
  const float* x       = (const float*)d_in[0];
  const float* g1      = (const float*)d_in[1];
  const float* b1      = (const float*)d_in[2];
  const float* w_qkv   = (const float*)d_in[3];
  const float* b_qkv   = (const float*)d_in[4];
  const float* rel_pos = (const float*)d_in[5];
  const float* w_proj  = (const float*)d_in[6];
  const float* b_proj  = (const float*)d_in[7];
  const float* g2      = (const float*)d_in[8];
  const float* b2      = (const float*)d_in[9];
  const float* w_mlp1  = (const float*)d_in[10];
  const float* b_mlp1  = (const float*)d_in[11];
  const float* w_mlp2  = (const float*)d_in[12];
  const float* b_mlp2  = (const float*)d_in[13];
  float* out = (float*)d_out;
  char* ws = (char*)d_ws;

  // Workspace layout (bytes). M1 reuses H/Q/K/Vt region (dead by then).
  const size_t OFF_WQKV  = 0;                       // 576*192*2   = 221184
  const size_t OFF_WPROJ = 221184;                  // 192*192*2   =  73728
  const size_t OFF_WM1   = 294912;                  // 768*192*2   = 294912
  const size_t OFF_WM2   = 589824;                  // 192*768*2   = 294912
  const size_t OFF_RELB  = 884736;                  // 6*64*64*4   =  98304
  const size_t OFF_H     = 1048576;                 // 65536*192*2 = 25165824
  const size_t OFF_Q     = OFF_H  + 25165824;
  const size_t OFF_K     = OFF_Q  + 25165824;
  const size_t OFF_VT    = OFF_K  + 25165824;
  const size_t OFF_ATTN  = OFF_H;                   // reuse (h dead after QKV)
  const size_t OFF_M1    = OFF_H;                   // reuse H..VT (100663296 B)
  const size_t OFF_X1    = OFF_VT + 25165824;       // 65536*192*4 = 50331648
  const size_t OFF_H2    = OFF_X1 + 50331648;       // 25165824  (total ~177 MB)

  __bf16* wqkvT  = (__bf16*)(ws + OFF_WQKV);
  __bf16* wprojT = (__bf16*)(ws + OFF_WPROJ);
  __bf16* wm1T   = (__bf16*)(ws + OFF_WM1);
  __bf16* wm2T   = (__bf16*)(ws + OFF_WM2);
  float*  relb   = (float*)(ws + OFF_RELB);
  __bf16* h      = (__bf16*)(ws + OFF_H);
  __bf16* Q      = (__bf16*)(ws + OFF_Q);
  __bf16* Kd     = (__bf16*)(ws + OFF_K);
  __bf16* Vt     = (__bf16*)(ws + OFF_VT);
  __bf16* attn   = (__bf16*)(ws + OFF_ATTN);
  __bf16* M1     = (__bf16*)(ws + OFF_M1);
  float*  x1     = (float*)(ws + OFF_X1);
  __bf16* h2     = (__bf16*)(ws + OFF_H2);

  // 1) weight transpose + bf16 convert
  k_transpose_bf16<<<(Cc*3*Cc + 255)/256, 256, 0, stream>>>(w_qkv,  wqkvT,  Cc,    3*Cc);
  k_transpose_bf16<<<(Cc*Cc   + 255)/256, 256, 0, stream>>>(w_proj, wprojT, Cc,    Cc);
  k_transpose_bf16<<<(Cc*4*Cc + 255)/256, 256, 0, stream>>>(w_mlp1, wm1T,   Cc,    4*Cc);
  k_transpose_bf16<<<(4*Cc*Cc + 255)/256, 256, 0, stream>>>(w_mlp2, wm2T,   4*Cc,  Cc);

  // 2) relative-position bias table
  k_relbias<<<(N_HEADS*PP*PP + 255)/256, 256, 0, stream>>>(rel_pos, relb);

  // 3) LN1 + roll + window partition
  k_ln1_window<<<RTOK/8, 256, 0, stream>>>(x, g1, b1, h);

  // 4) QKV GEMM  (M=65536, K=192, N=576)
  k_gemm_qkv<<<dim3(RTOK/128, (3*Cc)/64), 256, 0, stream>>>(h, wqkvT, b_qkv, Q, Kd, Vt);

  // 5) windowed attention  (1024 windows x 6 heads)
  k_attn<<<dim3(Bb*NW, N_HEADS), 128, 0, stream>>>(Q, Kd, Vt, relb, attn);

  // 6) proj GEMM + inverse window/roll + residual -> x1
  k_gemm_proj<<<dim3(RTOK/128, Cc/64), 256, 0, stream>>>(attn, wprojT, b_proj, x, x1);

  // 7) LN2
  k_ln2<<<RTOK/8, 256, 0, stream>>>(x1, g2, b2, h2);

  // 8) MLP1 GEMM + GELU  (N=768)
  k_gemm_mlp1<<<dim3(RTOK/128, (4*Cc)/64), 256, 0, stream>>>(h2, wm1T, b_mlp1, M1);

  // 9) MLP2 GEMM + residual -> out  (K=768, N=192)
  k_gemm_mlp2<<<dim3(RTOK/128, Cc/64), 256, 0, stream>>>(M1, wm2T, b_mlp2, x1, out);

  (void)in_sizes; (void)n_in; (void)out_size; (void)ws_size;
}
